// GraphConvBlock_79577154060686
// MI455X (gfx1250) — compile-verified, matrix operand order checked by reading, not run on hardware
//
#include <hip/hip_runtime.h>
#include <hip/hip_bf16.h>
#include <math.h>

// ---------------------------------------------------------------------------
// GATv2 x3 + BN + global max pool, fused pipeline for gfx1250 (MI455X).
// Node GEMMs run on v_wmma_f32_16x16x32_f16 (f32 accumulate); the per-layer
// We (16x64) tile is staged into LDS with the Tensor Data Mover.
// Problem constants (from reference):
#define NN     50000
#define EE     800000
#define DD     64
#define EDD    16
#define GG     32
#define NEGSL  0.2f
#define EPSBN  1e-5f
#define ETOT   (EE + NN)          // edges + self loops
#define NF     (NN * DD)          // node-feature elements

typedef __attribute__((ext_vector_type(16))) _Float16     v16h;
typedef __attribute__((ext_vector_type(8)))  float        v8f;
typedef __attribute__((ext_vector_type(4)))  unsigned int v4u;
typedef __attribute__((ext_vector_type(8)))  int          v8i;
typedef __attribute__((ext_vector_type(4)))  int          v4i;

#if defined(__has_builtin)
#if __has_builtin(__builtin_amdgcn_tensor_load_to_lds) && \
    __has_builtin(__builtin_amdgcn_s_wait_tensorcnt)
#define USE_TDM 1
#endif
#endif

// ---- order-preserving float <-> uint encoding for atomic max ---------------
__device__ __forceinline__ unsigned encf(float x) {
  unsigned u = __float_as_uint(x);
  return (u & 0x80000000u) ? ~u : (u | 0x80000000u);
}
__device__ __forceinline__ float decf(unsigned u) {
  return (u & 0x80000000u) ? __uint_as_float(u & 0x7FFFFFFFu)
                           : __uint_as_float(~u);
}
__device__ __forceinline__ float lrelu(float x) {
  return (x > 0.f) ? x : NEGSL * x;
}

// ---------------------------------------------------------------------------
// Self-loop edge_attr precompute: cnt[d] and sum of edge_attr per dst.
__global__ void k_zero_lattr(float* lattr, float* cnt) {
  int i = blockIdx.x * 256 + threadIdx.x;
  if (i < NN * EDD) lattr[i] = 0.f;
  if (i < NN)       cnt[i]   = 0.f;
}
__global__ void k_loop_sum(const int* __restrict__ ei,
                           const float* __restrict__ eattr,
                           float* lattr, float* cnt) {
  int e = blockIdx.x * 256 + threadIdx.x;
  if (e >= EE) return;
  int d = ei[EE + e];
  atomicAdd(&cnt[d], 1.f);
  const float* ea = eattr + e * EDD;
  #pragma unroll
  for (int k = 0; k < EDD; ++k) atomicAdd(&lattr[d * EDD + k], ea[k]);
}
__global__ void k_loop_div(float* lattr, const float* __restrict__ cnt) {
  int i = blockIdx.x * 256 + threadIdx.x;
  if (i >= NN * EDD) return;
  lattr[i] = lattr[i] / fmaxf(cnt[i >> 4], 1.f);
}

// ---------------------------------------------------------------------------
// WMMA GEMM: out[L] = hin @ Wl   (and optionally out[R] = hin @ Wr, sharing A)
// hin: N x 64 f32 row-major, W: 64 x 64 f32 row-major. N = 3125 * 16 exact.
// block = 128 (4 waves); block computes 16 rows x 64 cols; wave w -> cols 16w.
__global__ void k_gemm64(const float* __restrict__ hin,
                         const float* __restrict__ Wl,
                         const float* __restrict__ Wr,
                         float* __restrict__ outL,
                         float* __restrict__ outR,
                         int doBoth) {
  const int lane = threadIdx.x & 31;
  const int wave = threadIdx.x >> 5;
  const int row0 = blockIdx.x * 16;
  const int c0   = wave * 16;
  const int mrow = lane & 15;        // A row / D column-lane
  const int hi   = lane >> 4;        // 0: K-block lo, 1: K-block +8
  const int kb   = hi * 8;
  const int ncol = lane & 15;        // B column

  v8f accL = {};
  v8f accR = {};
  #pragma unroll
  for (int kc = 0; kc < DD; kc += 32) {
    // A fragment: lane holds row `mrow`, K = kc+kb+{0..7} and kc+16+kb+{0..7}
    v16h a;
    const float* arow = hin + (row0 + mrow) * DD + kc + kb;
    #pragma unroll
    for (int j = 0; j < 8; ++j) {
      a[j]     = (_Float16)arow[j];
      a[8 + j] = (_Float16)arow[16 + j];
    }
    // B fragment: lane holds column c0+ncol, same K striping as A.
    v16h bl;
    #pragma unroll
    for (int j = 0; j < 8; ++j) {
      int k1 = kc + kb + j, k2 = kc + 16 + kb + j;
      bl[j]     = (_Float16)Wl[k1 * DD + c0 + ncol];
      bl[8 + j] = (_Float16)Wl[k2 * DD + c0 + ncol];
    }
    accL = __builtin_amdgcn_wmma_f32_16x16x32_f16(
        false, a, false, bl, (short)0, accL, false, false);
    if (doBoth) {                       // uniform branch: EXEC stays all-ones
      v16h br;
      #pragma unroll
      for (int j = 0; j < 8; ++j) {
        int k1 = kc + kb + j, k2 = kc + 16 + kb + j;
        br[j]     = (_Float16)Wr[k1 * DD + c0 + ncol];
        br[8 + j] = (_Float16)Wr[k2 * DD + c0 + ncol];
      }
      accR = __builtin_amdgcn_wmma_f32_16x16x32_f16(
          false, a, false, br, (short)0, accR, false, false);
    }
  }
  // D layout: VGPR v -> row v (lanes 0-15) / row v+8 (lanes 16-31), col = ncol
  #pragma unroll
  for (int v = 0; v < 8; ++v) {
    int r = row0 + (hi ? v + 8 : v);
    outL[r * DD + c0 + ncol] = accL[v];
  }
  if (doBoth) {
    #pragma unroll
    for (int v = 0; v < 8; ++v) {
      int r = row0 + (hi ? v + 8 : v);
      outR[r * DD + c0 + ncol] = accR[v];
    }
  }
}

// ---------------------------------------------------------------------------
// Per-layer init: acc = bias (aggregation target), mx = enc(-inf-ish), den=0,
// BN partial sums zeroed.
__global__ void k_init(float* acc, const float* __restrict__ bias,
                       unsigned* mxe, float* den, float* colsum, float* colsq) {
  int i = blockIdx.x * 256 + threadIdx.x;
  if (i >= NF) return;
  int c = i & 63;
  acc[i] = bias[c];
  if (c == 0) { mxe[i >> 6] = 0u; den[i >> 6] = 0.f; }
  if (i < DD) { colsum[i] = 0.f; colsq[i] = 0.f; }
}

// ---------------------------------------------------------------------------
// Pass A: one wave per (edge or self-loop).  logit = att . lrelu(xl[s]+xr[d]+ea@We)
// and atomic segment-max into mxe[d].  We tile staged into LDS via TDM.
__global__ void k_edge_logits(const int* __restrict__ ei,
                              const float* __restrict__ eattr,
                              const float* __restrict__ lattr,
                              const float* __restrict__ xl,
                              const float* __restrict__ xr,
                              const float* __restrict__ We,
                              const float* __restrict__ att,
                              float* __restrict__ logit,
                              unsigned* __restrict__ mxe) {
  __shared__ float sWe[EDD * DD];
  __shared__ float sAtt[DD];
  int tid = threadIdx.x;
  if (tid < DD) sAtt[tid] = att[tid];
#ifdef USE_TDM
  if (tid == 0) {
    // Tensor DMA: 2-D tile (16 rows x 64 cols, 4B elems) global -> LDS.
    unsigned lds = (unsigned)(uintptr_t)(void*)sWe;  // low 32b = LDS byte addr
    unsigned long long ga = (unsigned long long)(uintptr_t)We;
    // D# group 0: count=1 | lds_addr | global_addr[56:0] | type=2
    v4u g0 = { 1u, lds, (unsigned)(ga & 0xFFFFFFFFu),
               (unsigned)(((ga >> 32) & 0x01FFFFFFu) | 0x80000000u) };
    // D# group 1: wg_mask=0, data_size=2 (4B); tensor_dim0=64 @bit48;
    // tensor_dim1=16 @bit80; tile_dim0=64 @bit112; tile_dim1=16 @bit128;
    // tile_dim2=0; tensor_dim0_stride=64 @bit160; tensor_dim1_stride=0.
    v8i g1 = { (int)0x00020000, (int)(64u << 16), (int)(16u << 16),
               (int)(64u << 16), 16, 64, 0, 0 };
    v4i g2 = { 0, 0, 0, 0 };          // dims 2..3 unused
    v4i g3 = { 0, 0, 0, 0 };          // dim 4 unused
#if __has_include(<hip/amd_detail/amd_gfx1250_TDM.h>)
    v8i g4 = { 0, 0, 0, 0, 0, 0, 0, 0 };
    __builtin_amdgcn_tensor_load_to_lds(g0, g1, g2, g3, g4, 0);
#else
    __builtin_amdgcn_tensor_load_to_lds(g0, g1, g2, g3, 0);
#endif
    __builtin_amdgcn_s_wait_tensorcnt(0);
  }
#else
  for (int i = tid; i < EDD * DD; i += 256) sWe[i] = We[i];
#endif
  __syncthreads();

  int e = blockIdx.x * 8 + (tid >> 5);
  if (e >= ETOT) return;
  int lane = tid & 31;
  int s, d;
  const float* ea;
  if (e < EE) { s = ei[e]; d = ei[EE + e]; ea = eattr + e * EDD; }
  else        { s = d = e - EE;            ea = lattr + (e - EE) * EDD; }
  __builtin_prefetch(ea, 0, 0);   // global_prefetch_b8

  int c1 = lane, c2 = lane + 32;
  float m1 = xl[s * DD + c1] + xr[d * DD + c1];
  float m2 = xl[s * DD + c2] + xr[d * DD + c2];
  #pragma unroll
  for (int k = 0; k < EDD; ++k) {
    float ev = ea[k];
    m1 += ev * sWe[k * DD + c1];
    m2 += ev * sWe[k * DD + c2];
  }
  float t = lrelu(m1) * sAtt[c1] + lrelu(m2) * sAtt[c2];
  #pragma unroll
  for (int off = 16; off > 0; off >>= 1) t += __shfl_xor(t, off, 32);
  if (lane == 0) {
    logit[e] = t;
    atomicMax(&mxe[d], encf(t));
  }
}

// Pass B: ex = exp(logit - mx[d]); den[d] += ex
__global__ void k_edge_exp(const int* __restrict__ ei,
                           const float* __restrict__ logit,
                           const unsigned* __restrict__ mxe,
                           float* __restrict__ exv,
                           float* __restrict__ den) {
  int e = blockIdx.x * 256 + threadIdx.x;
  if (e >= ETOT) return;
  int d = (e < EE) ? ei[EE + e] : (e - EE);
  float v = __expf(logit[e] - decf(mxe[d]));
  exv[e] = v;
  atomicAdd(&den[d], v);
}

// Pass C: acc[d] += (ex/den[d]) * xl[s]   (one wave per edge, 2 feats/lane)
__global__ void k_edge_aggregate(const int* __restrict__ ei,
                                 const float* __restrict__ exv,
                                 const float* __restrict__ den,
                                 const float* __restrict__ xl,
                                 float* __restrict__ acc) {
  int e = blockIdx.x * 8 + (threadIdx.x >> 5);
  if (e >= ETOT) return;
  int lane = threadIdx.x & 31;
  int s, d;
  if (e < EE) { s = ei[e]; d = ei[EE + e]; }
  else        { s = d = e - EE; }
  float alpha = exv[e] / den[d];
  int c1 = lane, c2 = lane + 32;
  atomicAdd(&acc[d * DD + c1], alpha * xl[s * DD + c1]);
  atomicAdd(&acc[d * DD + c2], alpha * xl[s * DD + c2]);
}

// ---------------------------------------------------------------------------
// BN stats over relu(acc): per-column sum / sumsq.
__global__ void k_bn_stats(const float* __restrict__ acc,
                           float* colsum, float* colsq) {
  int t = blockIdx.x * blockDim.x + threadIdx.x;
  int c = t & 63;
  int stride = (gridDim.x * blockDim.x) >> 6;
  float s1 = 0.f, s2 = 0.f;
  for (int r = t >> 6; r < NN; r += stride) {
    float v = fmaxf(acc[r * DD + c], 0.f);
    s1 += v; s2 += v * v;
  }
  atomicAdd(&colsum[c], s1);
  atomicAdd(&colsq[c], s2);
}

// BN apply: h = g*(relu(acc)-mean)*rsqrt(var+eps)+beta
__global__ void k_bn_apply(const float* __restrict__ acc,
                           const float* __restrict__ colsum,
                           const float* __restrict__ colsq,
                           const float* __restrict__ gam,
                           const float* __restrict__ bet,
                           float* __restrict__ hout) {
  int i = blockIdx.x * 256 + threadIdx.x;
  if (i >= NF) return;
  int c = i & 63;
  const float invN = 1.f / (float)NN;
  float mean = colsum[c] * invN;
  float var  = colsq[c] * invN - mean * mean;   // biased, as reference
  float v = fmaxf(acc[i], 0.f);
  hout[i] = gam[c] * (v - mean) * rsqrtf(var + EPSBN) + bet[c];
}

// ---------------------------------------------------------------------------
// Global max pool over batch ids.
__global__ void k_pool_init(unsigned* pool) {
  int i = blockIdx.x * 256 + threadIdx.x;
  if (i < GG * DD) pool[i] = 0u;
}
__global__ void k_pool(const float* __restrict__ h,
                       const int* __restrict__ batch, unsigned* pool) {
  int i = blockIdx.x * 256 + threadIdx.x;
  if (i >= NF) return;
  int node = i >> 6, c = i & 63;
  atomicMax(&pool[batch[node] * DD + c], encf(h[i]));
}
__global__ void k_pool_out(const unsigned* __restrict__ pool, float* out) {
  int i = blockIdx.x * 256 + threadIdx.x;
  if (i < GG * DD) out[i] = decf(pool[i]);
}

// ---------------------------------------------------------------------------
extern "C" void kernel_launch(void* const* d_in, const int* in_sizes, int n_in,
                              void* d_out, int out_size, void* d_ws, size_t ws_size,
                              hipStream_t stream) {
  const float* x     = (const float*)d_in[0];
  const int*   ei    = (const int*)  d_in[1];   // [2,E]: src then dst
  const float* eattr = (const float*)d_in[2];
  const int*   batch = (const int*)  d_in[3];
  const float* wl0   = (const float*)d_in[4];
  const float* wr0   = (const float*)d_in[5];
  const float* we0   = (const float*)d_in[6];
  const float* att0  = (const float*)d_in[7];
  const float* b0    = (const float*)d_in[8];
  const float* g0    = (const float*)d_in[9];
  const float* bt0   = (const float*)d_in[10];
  const float* W_s   = (const float*)d_in[11];  // [2,64,64]
  const float* We_s  = (const float*)d_in[12];  // [2,16,64]
  const float* att_s = (const float*)d_in[13];  // [2,64]
  const float* b_s   = (const float*)d_in[14];
  const float* g_s   = (const float*)d_in[15];
  const float* bt_s  = (const float*)d_in[16];

  // Workspace carve-up (floats). Total ~15.46M floats (~62 MB).
  float* base   = (float*)d_ws;
  float* xl     = base;                       // N*64
  float* xr     = xl + NF;                    // N*64
  float* acc    = xr + NF;                    // N*64
  float* hbuf   = acc + NF;                   // N*64
  float* logit  = hbuf + NF;                  // E+N
  float* exv    = logit + ETOT;               // E+N
  unsigned* mxe = (unsigned*)(exv + ETOT);    // N
  float* den    = (float*)(mxe + NN);         // N
  float* cnt    = den + NN;                   // N
  float* lattr  = cnt + NN;                   // N*16
  float* colsum = lattr + NN * EDD;           // 64
  float* colsq  = colsum + DD;                // 64
  unsigned* pool = (unsigned*)(colsq + DD);   // G*64

  const int B = 256;
  // --- self-loop attr = mean of incoming edge_attr ---
  k_zero_lattr<<<(NN * EDD + B - 1) / B, B, 0, stream>>>(lattr, cnt);
  k_loop_sum  <<<(EE + B - 1) / B,       B, 0, stream>>>(ei, eattr, lattr, cnt);
  k_loop_div  <<<(NN * EDD + B - 1) / B, B, 0, stream>>>(lattr, cnt);

  for (int layer = 0; layer < 3; ++layer) {
    const float *hin, *Wl, *Wr, *Wep, *attp, *bp, *gp, *btp;
    int doBoth;
    if (layer == 0) {
      hin = x;   Wl = wl0; Wr = wr0; Wep = we0; attp = att0;
      bp = b0; gp = g0; btp = bt0; doBoth = 1;
    } else {
      int i = layer - 1;
      hin = hbuf;
      Wl = W_s  + i * DD * DD;  Wr = Wl;           // share_weights=True
      Wep = We_s + i * EDD * DD;
      attp = att_s + i * DD; bp = b_s + i * DD;
      gp = g_s + i * DD;     btp = bt_s + i * DD;
      doBoth = 0;
    }
    float* xrp = doBoth ? xr : xl;

    k_gemm64<<<NN / 16, 128, 0, stream>>>(hin, Wl, Wr, xl, xr, doBoth);
    k_init  <<<NF / B, B, 0, stream>>>(acc, bp, mxe, den, colsum, colsq);
    k_edge_logits   <<<ETOT / 8, B, 0, stream>>>(ei, eattr, lattr, xl, xrp,
                                                 Wep, attp, logit, mxe);
    k_edge_exp      <<<(ETOT + B - 1) / B, B, 0, stream>>>(ei, logit, mxe, exv, den);
    k_edge_aggregate<<<ETOT / 8, B, 0, stream>>>(ei, exv, den, xl, acc);
    k_bn_stats<<<1024, B, 0, stream>>>(acc, colsum, colsq);
    k_bn_apply<<<NF / B, B, 0, stream>>>(acc, colsum, colsq, gp, btp, hbuf);
  }

  k_pool_init<<<(GG * DD + B - 1) / B, B, 0, stream>>>(pool);
  k_pool     <<<NF / B, B, 0, stream>>>(hbuf, batch, pool);
  k_pool_out <<<(GG * DD + B - 1) / B, B, 0, stream>>>(pool, (float*)d_out);
}
